// MultiHeadedAttention_4183298146680
// MI455X (gfx1250) — matrix operand order repaired and checked by reading, not compiled, not run on hardware
//
#include <hip/hip_runtime.h>
#include <hip/hip_bf16.h>

#define B_   2
#define S_   4096
#define HID_ 512
#define H_   8
#define DK_  64

#define INV_LOG_DK 0.24044917348149386f   // 1/ln(64)
#define LOG2E      1.4426950408889634f

typedef __bf16 bf16_t;
typedef __attribute__((ext_vector_type(16))) __bf16 v16bf;
typedef __attribute__((ext_vector_type(8)))  __bf16 v8bf;
typedef __attribute__((ext_vector_type(8)))  float  v8f;
typedef __attribute__((ext_vector_type(4)))  float  v4f;

// 16-byte int vector, matching the async-LDS builtin's parameter type
// (diagnostic: param 1 is AS1 pointer to __attribute__((vector_size(16))) int).
typedef int v4i __attribute__((__vector_size__(16)));
typedef __attribute__((address_space(1))) v4i g_v4i;
typedef __attribute__((address_space(3))) v4i l_v4i;

#if defined(__has_builtin)
#  if __has_builtin(__builtin_amdgcn_global_load_async_to_lds_b128)
#    define HAVE_ASYNC_BUILTIN 1
#  endif
#  if __has_builtin(__builtin_amdgcn_s_wait_asynccnt)
#    define HAVE_WAIT_ASYNC_BUILTIN 1
#  endif
#endif

// Async 16B copy global->LDS (tracked by ASYNCcnt, not LOADcnt).
__device__ __forceinline__ void async_cp_b128(bf16_t* lds_dst, const bf16_t* gsrc) {
#if defined(HAVE_ASYNC_BUILTIN)
  __builtin_amdgcn_global_load_async_to_lds_b128(
      (g_v4i*)(unsigned long long)gsrc,
      (l_v4i*)(unsigned)(unsigned long long)lds_dst,   // low 32 bits = LDS offset
      0, 0);
#else
  unsigned loff = (unsigned)(unsigned long long)lds_dst;
  asm volatile("global_load_async_to_lds_b128 %0, %1, off"
               :: "v"(loff), "v"(gsrc) : "memory");
#endif
}

__device__ __forceinline__ void wait_async0() {
#if defined(HAVE_WAIT_ASYNC_BUILTIN)
  __builtin_amdgcn_s_wait_asynccnt(0);
#else
  asm volatile("s_wait_asynccnt 0x0" ::: "memory");
#endif
}

__device__ __forceinline__ v8f zero8() {
  v8f r;
#pragma unroll
  for (int i = 0; i < 8; ++i) r[i] = 0.0f;
  return r;
}

__device__ __forceinline__ v8f wmma_bf16(v16bf a, v16bf b, v8f c) {
  // D = A(16x32 bf16) * B(32x16 bf16) + C(16x16 f32)
  return __builtin_amdgcn_wmma_f32_16x16x32_bf16(false, a, false, b,
                                                 (short)0, c, false, false);
}

__device__ __forceinline__ v16bf cat16(v8bf lo, v8bf hi) {
  v16bf r;
#pragma unroll
  for (int i = 0; i < 8; ++i) { r[i] = lo[i]; r[i + 8] = hi[i]; }
  return r;
}

__device__ __forceinline__ v16bf cvt16(v4f a0, v4f a1, v4f a2, v4f a3) {
  v16bf r;
#pragma unroll
  for (int i = 0; i < 4; ++i) {
    r[i]      = (__bf16)a0[i];
    r[4 + i]  = (__bf16)a1[i];
    r[8 + i]  = (__bf16)a2[i];
    r[12 + i] = (__bf16)a3[i];
  }
  return r;
}

// ---------------------------------------------------------------------------
// QKV projection: out[m][n] = sum_d X[m][d] * W[n][d]; M=B*S, N=K=HID.
// One 16x16 tile per wave, 16x (16x16x32 bf16 WMMA) K-loop.
// A lane layout (16x32, MxK): m = lane&15, elem e -> k = (e<8 ? e : e+8) + 8*(lane>>4)
// B lane layout (32x16, KxN): n = lane&15, elem e -> k = (e<8 ? e : e+8) + 8*(lane>>4)
// transpose_v=0: store [b,h,s,dk] bf16 ; =1: store [b,h,dk,s] bf16 (for V).
// ---------------------------------------------------------------------------
__global__ __launch_bounds__(256) void proj_qkv_kernel(
    const float* __restrict__ X, const float* __restrict__ W,
    bf16_t* __restrict__ out, int transpose_v)
{
  const int lane = threadIdx.x & 31;
  const int wave = blockIdx.x * (blockDim.x >> 5) + (threadIdx.x >> 5);
  const int M_TILES = (B_ * S_) / 16;            // 512
  const int mt = wave % M_TILES;
  const int nt = wave / M_TILES;                 // 0..31
  const int m0 = mt * 16, n0 = nt * 16;
  const int row = lane & 15, half = lane >> 4;

  v8f acc = zero8();
  const float* arow = X + (size_t)(m0 + row) * HID_;
  const float* brow = W + (size_t)(n0 + row) * HID_;
#pragma unroll 4
  for (int kb = 0; kb < HID_; kb += 32) {
    const float* ap = arow + kb + 8 * half;
    v16bf a = cvt16(*(const v4f*)(ap),      *(const v4f*)(ap + 4),
                    *(const v4f*)(ap + 16), *(const v4f*)(ap + 20));
    const float* bp = brow + kb + 8 * half;
    v16bf b = cvt16(*(const v4f*)(bp),     *(const v4f*)(bp + 4),
                    *(const v4f*)(bp + 16), *(const v4f*)(bp + 20));
    acc = wmma_bf16(a, b, acc);
  }

  // C/D layout: reg r holds (m = m0 + r + 8*half, n = n0 + row)
  const int n_glob = n0 + row;
  const int h  = n_glob >> 6;
  const int dk = n_glob & 63;
  const int bidx   = m0 / S_;
  const int s_base = m0 % S_;
  if (transpose_v) {
    v8bf pk;
#pragma unroll
    for (int r = 0; r < 8; ++r) pk[r] = (__bf16)acc[r];
    bf16_t* dst = out + ((size_t)(bidx * H_ + h) * DK_ + dk) * S_
                      + s_base + 8 * half;
    *(v8bf*)dst = pk;                // 8 consecutive s positions, one 16B store
  } else {
    bf16_t* dst = out + ((size_t)(bidx * H_ + h) * S_ + s_base + 8 * half) * DK_ + dk;
#pragma unroll
    for (int r = 0; r < 8; ++r) dst[(size_t)r * DK_] = (__bf16)acc[r];
  }
}

// ---------------------------------------------------------------------------
// Flash attention. One block = 8 waves = 8 consecutive 16-row q-tiles of ONE
// (b,h). Keys streamed in chunks of 32; each chunk's K (32x64) and V (64x32,
// pre-transposed) staged ONCE per block into LDS with async global->LDS
// copies, double-buffered so the next chunk's DMA overlaps this chunk's
// WMMA + softmax. Fragments then come from LDS (ds_load_b128).
// ---------------------------------------------------------------------------
__global__ __launch_bounds__(256) void flash_attn_kernel(
    const bf16_t* __restrict__ Q, const bf16_t* __restrict__ K,
    const bf16_t* __restrict__ Vt, const int* __restrict__ mask,
    bf16_t* __restrict__ Xo)
{
  __shared__ __align__(16) bf16_t Ks[2][32 * 64];   // 2 x 4 KB
  __shared__ __align__(16) bf16_t Vs[2][64 * 32];   // 2 x 4 KB
  __shared__ __align__(16) bf16_t pbuf[8 * 512];    // 8 KB (1 KB / wave)

  const int tid  = threadIdx.x;
  const int lane = tid & 31;
  const int wid  = tid >> 5;
  const int bh   = blockIdx.x >> 5;                    // 32 blocks per (b,h)
  const int qt   = (blockIdx.x * 8 + wid) & 255;
  const int bidx = bh >> 3;
  const int h    = bh & 7;
  const int q0   = qt * 16;
  const int row = lane & 15, half = lane >> 4;

  const size_t bh_off = (size_t)bh * S_ * DK_;
  const bf16_t* Qbh = Q  + bh_off;
  const bf16_t* Kbh = K  + bh_off;                     // [S_][DK_]
  const bf16_t* Vbh = Vt + bh_off;                     // [DK_][S_]
  const int* mb = mask + bidx * S_;

  // staging assignment: K chunk is one contiguous 4KB blob (rows contiguous);
  // V chunk is 64 rows x 64B. 256 threads x 16B each.
  const int vr = tid >> 2;          // 0..63
  const int vc = (tid & 3) * 8;     // 0,8,16,24

  // Q A-fragments (d-chunks 0..31, 32..63), loaded once.
  const bf16_t* qrow = Qbh + (size_t)(q0 + row) * DK_;
  v16bf aq0 = cat16(*(const v8bf*)(qrow +      16 * half),
                    *(const v8bf*)(qrow + 8 + 16 * half));
  v16bf aq1 = cat16(*(const v8bf*)(qrow + 32 + 16 * half),
                    *(const v8bf*)(qrow + 40 + 16 * half));

  float mi[8], li[8];
#pragma unroll
  for (int r = 0; r < 8; ++r) { mi[r] = -3.0e38f; li[r] = 0.0f; }
  v8f o[4] = { zero8(), zero8(), zero8(), zero8() };

  bf16_t* pb = pbuf + wid * 512;

  // prologue: stage chunk 0 into buffer 0
  async_cp_b128(&Ks[0][tid * 8], Kbh + (size_t)tid * 8);
  async_cp_b128(&Vs[0][vr * 32 + vc], Vbh + (size_t)vr * S_ + vc);

  for (int kc = 0; kc < S_; kc += 32) {
    const int buf = (kc >> 5) & 1;
    wait_async0();          // my staging (and my prefetch) complete
    __syncthreads();        // everyone's staging complete -> buffer readable

    // prefetch next chunk into the other buffer (overlaps compute below)
    if (kc + 32 < S_) {
      const int kn = kc + 32;
      async_cp_b128(&Ks[buf ^ 1][tid * 8], Kbh + (size_t)kn * DK_ + tid * 8);
      async_cp_b128(&Vs[buf ^ 1][vr * 32 + vc],
                    Vbh + (size_t)vr * S_ + kn + vc);
    }

    // ---- scores 16x32 = Q (16x64) * K^T, K fragments from LDS ----
    v8f s0 = zero8(), s1 = zero8();
    {
      const bf16_t* k0 = &Ks[buf][(size_t)row * DK_ + 16 * half];
      v16bf bk00 = cat16(*(const v8bf*)(k0),      *(const v8bf*)(k0 + 8));
      v16bf bk01 = cat16(*(const v8bf*)(k0 + 32), *(const v8bf*)(k0 + 40));
      s0 = wmma_bf16(aq0, bk00, s0);
      s0 = wmma_bf16(aq1, bk01, s0);
      const bf16_t* k1 = &Ks[buf][(size_t)(16 + row) * DK_ + 16 * half];
      v16bf bk10 = cat16(*(const v8bf*)(k1),      *(const v8bf*)(k1 + 8));
      v16bf bk11 = cat16(*(const v8bf*)(k1 + 32), *(const v8bf*)(k1 + 40));
      s1 = wmma_bf16(aq0, bk10, s1);
      s1 = wmma_bf16(aq1, bk11, s1);
    }

    // ---- scale + key mask ----
    const float msk0 = (mb[kc + row]      == 0) ? -1.0e30f : 0.0f;
    const float msk1 = (mb[kc + 16 + row] == 0) ? -1.0e30f : 0.0f;
#pragma unroll
    for (int r = 0; r < 8; ++r) {
      s0[r] = s0[r] * INV_LOG_DK + msk0;
      s1[r] = s1[r] * INV_LOG_DK + msk1;
    }

    // ---- online softmax (row m lives in 16 lanes of one half-wave) ----
    float p0[8], p1[8];
#pragma unroll
    for (int r = 0; r < 8; ++r) {
      float rm = fmaxf(s0[r], s1[r]);
      rm = fmaxf(rm, __shfl_xor(rm, 1, 32));
      rm = fmaxf(rm, __shfl_xor(rm, 2, 32));
      rm = fmaxf(rm, __shfl_xor(rm, 4, 32));
      rm = fmaxf(rm, __shfl_xor(rm, 8, 32));
      const float mnew = fmaxf(mi[r], rm);
      const float corr = exp2f((mi[r] - mnew) * LOG2E);
      const float e0 = exp2f((s0[r] - mnew) * LOG2E);
      const float e1 = exp2f((s1[r] - mnew) * LOG2E);
      float rs = e0 + e1;
      rs += __shfl_xor(rs, 1, 32);
      rs += __shfl_xor(rs, 2, 32);
      rs += __shfl_xor(rs, 4, 32);
      rs += __shfl_xor(rs, 8, 32);
      li[r] = li[r] * corr + rs;
      mi[r] = mnew;
      o[0][r] *= corr; o[1][r] *= corr; o[2][r] *= corr; o[3][r] *= corr;
      p0[r] = e0; p1[r] = e1;
    }

    // ---- re-layout P: C-layout regs -> LDS [m][k_local] -> A-fragment ----
#pragma unroll
    for (int r = 0; r < 8; ++r) {
      pb[(r + 8 * half) * 32 + row]      = (__bf16)p0[r];
      pb[(r + 8 * half) * 32 + 16 + row] = (__bf16)p1[r];
    }
    __syncthreads();     // uniform trip count; orders pbuf store->load
    const bf16_t* prow = pb + row * 32 + 16 * half;
    v16bf ap = cat16(*(const v8bf*)(prow), *(const v8bf*)(prow + 8));

    // ---- O += P (16x32) * V (32x64), V fragments from LDS ----
#pragma unroll
    for (int dt = 0; dt < 4; ++dt) {
      const bf16_t* vrow = &Vs[buf][(size_t)(dt * 16 + row) * 32 + 16 * half];
      v16bf bv = cat16(*(const v8bf*)(vrow), *(const v8bf*)(vrow + 8));
      o[dt] = wmma_bf16(ap, bv, o[dt]);
    }
    // buffer is refilled by the next iteration's prefetch -> protected by its
    // top-of-loop __syncthreads(); no end-of-loop barrier needed.
  }

  // ---- normalize and store to merged-head layout [b,s,hid] ----
  float inv[8];
#pragma unroll
  for (int r = 0; r < 8; ++r) inv[r] = 1.0f / li[r];
#pragma unroll
  for (int dt = 0; dt < 4; ++dt)
#pragma unroll
    for (int r = 0; r < 8; ++r) {
      size_t addr = ((size_t)bidx * S_ + q0 + r + 8 * half) * HID_
                    + h * DK_ + dt * 16 + row;
      Xo[addr] = (__bf16)(o[dt][r] * inv[r]);
    }
}

// ---------------------------------------------------------------------------
// Output projection: out[m][n] = sum_d Xi[m][d] * Wo[n][d]; fp32 output.
// ---------------------------------------------------------------------------
__global__ __launch_bounds__(256) void out_proj_kernel(
    const bf16_t* __restrict__ Xi, const float* __restrict__ Wo,
    float* __restrict__ out)
{
  const int lane = threadIdx.x & 31;
  const int wave = blockIdx.x * (blockDim.x >> 5) + (threadIdx.x >> 5);
  const int M_TILES = (B_ * S_) / 16;
  const int mt = wave % M_TILES;
  const int nt = wave / M_TILES;
  const int m0 = mt * 16, n0 = nt * 16;
  const int row = lane & 15, half = lane >> 4;

  v8f acc = zero8();
  const bf16_t* arow = Xi + (size_t)(m0 + row) * HID_;
  const float*  brow = Wo + (size_t)(n0 + row) * HID_;
#pragma unroll 4
  for (int kb = 0; kb < HID_; kb += 32) {
    const bf16_t* ap = arow + kb + 8 * half;
    v16bf a = cat16(*(const v8bf*)(ap), *(const v8bf*)(ap + 16));
    const float* bp = brow + kb + 8 * half;
    v16bf b = cvt16(*(const v4f*)(bp),     *(const v4f*)(bp + 4),
                    *(const v4f*)(bp + 16), *(const v4f*)(bp + 20));
    acc = wmma_bf16(a, b, acc);
  }
#pragma unroll
  for (int r = 0; r < 8; ++r)
    out[(size_t)(m0 + r + 8 * half) * HID_ + n0 + row] = acc[r];
}

// ---------------------------------------------------------------------------
extern "C" void kernel_launch(void* const* d_in, const int* in_sizes, int n_in,
                              void* d_out, int out_size, void* d_ws, size_t ws_size,
                              hipStream_t stream) {
  (void)in_sizes; (void)n_in; (void)out_size; (void)ws_size;
  const float* query = (const float*)d_in[0];
  const float* key   = (const float*)d_in[1];
  const float* value = (const float*)d_in[2];
  const int*   mask  = (const int*)d_in[3];
  const float* Wq    = (const float*)d_in[4];
  const float* Wk    = (const float*)d_in[5];
  const float* Wv    = (const float*)d_in[6];
  const float* Wo    = (const float*)d_in[7];
  float* out = (float*)d_out;

  // Workspace carve: q, k, v (per-head bf16) + merged attention output (bf16)
  const size_t per = (size_t)B_ * H_ * S_ * DK_;     // 4Mi elems = 8 MB each
  bf16_t* qb = (bf16_t*)d_ws;
  bf16_t* kb = qb + per;
  bf16_t* vb = kb + per;
  bf16_t* xb = vb + per;

  dim3 blk(256);
  dim3 grid_gemm(2048);   // 16384 waves = 512 M-tiles * 32 N-tiles

  proj_qkv_kernel<<<grid_gemm, blk, 0, stream>>>(query, Wq, qb, 0);
  proj_qkv_kernel<<<grid_gemm, blk, 0, stream>>>(key,   Wk, kb, 0);
  proj_qkv_kernel<<<grid_gemm, blk, 0, stream>>>(value, Wv, vb, 1);

  flash_attn_kernel<<<dim3(512), blk, 0, stream>>>(qb, kb, vb, mask, xb);

  out_proj_kernel<<<grid_gemm, blk, 0, stream>>>(xb, Wo, out);
}
